// AttentionRNN_30107720745169
// MI455X (gfx1250) — compile-verified, matrix operand order checked by reading, not run on hardware
//
#include <hip/hip_runtime.h>
#include <stdint.h>

#define L 32
#define F 28
#define H 512
#define BATCH 4096
#define G 16          // persistent workgroups in the scan kernel
#define NT_ALL 128    // 2048/16 n-tiles
#define KT 16         // 512/32 k-steps

typedef __attribute__((ext_vector_type(16))) __bf16 v16bf;
typedef __attribute__((ext_vector_type(8)))  float  v8f;

union FragU { uint4 q[2]; v16bf v; };

__device__ __forceinline__ unsigned short f2bf(float f) {
  unsigned u = __builtin_bit_cast(unsigned, f);
  unsigned r = u + 0x7FFFu + ((u >> 16) & 1u);   // round-to-nearest-even
  return (unsigned short)(r >> 16);
}
__device__ __forceinline__ float bf2f(unsigned short h) {
  unsigned u = ((unsigned)h) << 16;
  return __builtin_bit_cast(float, u);
}
__device__ __forceinline__ float sigmoidf_(float x) { return 1.0f / (1.0f + __expf(-x)); }
// WMMA 16-bit A-fragment K-pair base for (vgpr v, lane-half)
__device__ __forceinline__ int fragK0(int v, int half) {
  return 2 * (v & 3) + 8 * half + 16 * (v >> 2);
}

// ---------------- conv1 residual + u_a (parallel over B) ----------------
__global__ void k_pre(const float* __restrict__ inp, const float* __restrict__ mask,
                      const float* __restrict__ w1, const float* __restrict__ b1,
                      const float* __restrict__ cw, const float* __restrict__ cb,
                      float* __restrict__ xF32, float* __restrict__ uA) {
  int b = blockIdx.x;
  __shared__ float sIn[L][F];
  __shared__ float sW[F * F * 3];
  __shared__ float sX[L][F];
  for (int t = threadIdx.x; t < L * F; t += blockDim.x) sIn[t / F][t % F] = inp[b * L * F + t];
  for (int t = threadIdx.x; t < F * F * 3; t += blockDim.x) sW[t] = w1[t];
  __syncthreads();
  for (int t = threadIdx.x; t < L * F; t += blockDim.x) {
    int l = t / F, fo = t % F;
    float acc = b1[fo];
    for (int fi = 0; fi < F; ++fi) {
      const float* wr = &sW[fo * F * 3 + fi * 3];
      float s = sIn[l][fi] * wr[1];
      if (l > 0)     s += sIn[l - 1][fi] * wr[0];
      if (l < L - 1) s += sIn[l + 1][fi] * wr[2];
      acc += s;
    }
    float xv = acc * mask[b * L + l];
    xv = (xv > 0.f) ? xv : (__expf(xv) - 1.f);   // ELU
    xv += sIn[l][fo];                            // residual
    sX[l][fo] = xv;
    xF32[b * L * F + t] = xv;
  }
  __syncthreads();
  if (threadIdx.x < F) {
    int f = threadIdx.x;
    float s = cb[0];
    for (int l = 0; l < L; ++l) s += sX[l][f] * cw[l];
    uA[b * F + f] = s;
  }
}

// ---------------- pack W_hh / W_ih into WMMA B-fragment order (bf16) ----------------
__global__ void k_pack(const float* __restrict__ Whh, const float* __restrict__ Wih,
                       unsigned* __restrict__ whhF, unsigned* __restrict__ wihF) {
  int blk = blockIdx.x, t = threadIdx.x;
  int lane = t >> 3, v = t & 7;
  int row = lane & 15, half = lane >> 4;
  int k0 = fragK0(v, half);
  if (blk < KT * NT_ALL) {
    int kt = blk >> 7, nt = blk & 127;
    const float* p = Whh + (nt * 16 + row) * H + kt * 32 + k0;  // W_hh[n][k] row-major == B^T
    whhF[blk * 256 + t] = (unsigned)f2bf(p[0]) | ((unsigned)f2bf(p[1]) << 16);
  } else {
    int nt = blk - KT * NT_ALL;
    const float* p = Wih + (nt * 16 + row) * L + k0;            // K = L = 32, single k-step
    wihF[nt * 256 + t] = (unsigned)f2bf(p[0]) | ((unsigned)f2bf(p[1]) << 16);
  }
}

// ---------------- zero recurrent state + step counter (every call: graph-safe) ----------------
__global__ void k_init(unsigned* __restrict__ hFrag, unsigned* __restrict__ hPlainW,
                       unsigned* __restrict__ ctr) {
  int i = blockIdx.x * blockDim.x + threadIdx.x;
  if (i < 2 * 2 * KT * 256) hFrag[i] = 0;
  if (i < 2 * L * H / 2)    hPlainW[i] = 0;    // 2 parities * 32*512 bf16 as dwords
  if (i == 0) *ctr = 0;
}

// ---------------- the sequential scan: 16 persistent WGs, WMMA GEMM per step ----------------
__global__ void __launch_bounds__(256, 1)
k_scan(const float* __restrict__ uA, const float* __restrict__ xF32,
       const float* __restrict__ biasMat, const float* __restrict__ fc1w,
       const float* __restrict__ fc1b, const unsigned* __restrict__ whhF,
       const unsigned* __restrict__ wihF, const float* __restrict__ bih,
       const float* __restrict__ bhh, const float* __restrict__ w2,
       unsigned* __restrict__ hFrag, unsigned short* __restrict__ hPlain,
       float* __restrict__ partial, unsigned* __restrict__ ctr) {
  const int g = blockIdx.x;          // 0..15: owns h columns [32g, 32g+32)
  const int t = threadIdx.x;
  const int lane = t & 31;           // wave32
  const int w = t >> 5;              // 8 waves
  const int q = w >> 1;              // gate (i,f,g,o)
  const int ntg = 32 * q + 2 * g + (w & 1);  // global n-tile of this wave

  // A-frag staging buffer overlaid with post-GEMM buffers (never coexist)
  __shared__ __align__(16) union SharedU {
    unsigned aFrag[2 * KT * 256];                       // 32 KB
    struct { float gates[4][L][L]; float hbuf[L][L]; } p3;  // 20 KB
  } U;
  __shared__ float    sX[L][F];
  __shared__ float    sAttn[L][F];
  __shared__ float    sWa[L];
  __shared__ float    sWaPart[L][8];
  __shared__ float    sCtx[L][L];
  __shared__ unsigned sCtxFrag[2][256];
  __shared__ float    sC[L][L];

  for (int i = t; i < L * L; i += 256) sC[i >> 5][i & 31] = 0.f;  // c0 = 0
  __syncthreads();

  for (int s = 0; s < BATCH; ++s) {
    const int b = s;
    const int par = s & 1;
    const unsigned short* hP = hPlain + par * (L * H);
    const unsigned*       hF = hFrag + par * (2 * KT * 256);

    // cooperative stage: this step's A-fragments (32 KB) -> LDS, one L2 round trip
    {
      const uint4* src = (const uint4*)hF;
      uint4* dst = (uint4*)U.aFrag;
      for (int i = t; i < 2 * KT * 64; i += 256) dst[i] = src[i];
    }
    // warm next batch element's attention inputs while we compute
    if (s + 1 < BATCH && t < 32) {
      __builtin_prefetch(xF32 + (b + 1) * L * F + t * 32, 0, 1);
      __builtin_prefetch(biasMat + (b + 1) * L * F + t * 32, 0, 1);
    }
    // w_a = h @ fc1_w^T + fc1_b  (32x8 partial tree)
    {
      int l = t >> 3, seg = t & 7;
      float acc = 0.f;
      for (int k = seg * 64; k < seg * 64 + 64; ++k) acc += bf2f(hP[l * H + k]) * fc1w[k];
      sWaPart[l][seg] = acc;
    }
    for (int i = t; i < L * F; i += 256) sX[i / F][i % F] = xF32[b * L * F + i];
    __syncthreads();
    if (t < L) {
      float a = fc1b[0];
      for (int e = 0; e < 8; ++e) a += sWaPart[t][e];
      sWa[t] = a;
    }
    __syncthreads();
    // scores = leaky_relu(u_b + w_a) + bias_mat
    for (int i = t; i < L * F; i += 256) {
      int l = i / F, f = i % F;
      float v = uA[b * F + f] + sWa[l];
      v = (v > 0.f) ? v : 0.01f * v;
      sAttn[l][f] = v + biasMat[b * L * F + i];
    }
    __syncthreads();
    // softmax over L (column-wise)
    if (t < F) {
      int f = t;
      float mx = -3.0e38f;
      for (int l = 0; l < L; ++l) mx = fmaxf(mx, sAttn[l][f]);
      float sum = 0.f;
      for (int l = 0; l < L; ++l) { float e = __expf(sAttn[l][f] - mx); sAttn[l][f] = e; sum += e; }
      float inv = 1.f / sum;
      for (int l = 0; l < L; ++l) sAttn[l][f] *= inv;
    }
    __syncthreads();
    // ctx = attn @ x^T  (32x32, K=28 -> VALU)
    for (int i = t; i < L * L; i += 256) {
      int l = i >> 5, j = i & 31;
      float acc = 0.f;
      for (int f = 0; f < F; ++f) acc += sAttn[l][f] * sX[j][f];
      sCtx[l][j] = acc;
    }
    __syncthreads();
    // pack ctx into A-fragments (bf16)
    for (int u = t; u < 2 * 256; u += 256) {
      int mt = u >> 8, r = u & 255;
      int ln = r >> 3, v = r & 7;
      int m = 16 * mt + (ln & 15);
      int k0 = fragK0(v, ln >> 4);
      sCtxFrag[mt][r] = (unsigned)f2bf(sCtx[m][k0]) | ((unsigned)f2bf(sCtx[m][k0 + 1]) << 16);
    }
    __syncthreads();   // aFrag staging + ctx frags complete

    // gates[32, 128 cols] = h @ W_hh^T + ctx @ W_ih^T
    // B-fragments (whhF/wihF) are loop-invariant -> compiler keeps them in VGPRs.
    // A-fragments come from LDS, double-buffered so ds_load(kt+1) overlaps wmma(kt).
    v8f acc0 = {0.f, 0.f, 0.f, 0.f, 0.f, 0.f, 0.f, 0.f};
    v8f acc1 = acc0;
    {
      const uint4* Abase = (const uint4*)U.aFrag;   // [mt*KT + kt]*64 + lane*2
      const int base0 = lane * 2;
      FragU a0[2], a1[2];
      a0[0].q[0] = Abase[base0];
      a0[0].q[1] = Abase[base0 + 1];
      a1[0].q[0] = Abase[KT * 64 + base0];
      a1[0].q[1] = Abase[KT * 64 + base0 + 1];
#pragma unroll
      for (int kt = 0; kt < KT; ++kt) {
        const int cur = kt & 1, nxt = cur ^ 1;
        if (kt + 1 < KT) {
          int o = (kt + 1) * 64 + base0;
          a0[nxt].q[0] = Abase[o];
          a0[nxt].q[1] = Abase[o + 1];
          a1[nxt].q[0] = Abase[KT * 64 + o];
          a1[nxt].q[1] = Abase[KT * 64 + o + 1];
        }
        FragU bb;
        const uint4* pb = (const uint4*)(whhF + (kt * NT_ALL + ntg) * 256 + lane * 8);
        bb.q[0] = pb[0]; bb.q[1] = pb[1];
        acc0 = __builtin_amdgcn_wmma_f32_16x16x32_bf16(false, a0[cur].v, false, bb.v, (short)0, acc0, false, false);
        acc1 = __builtin_amdgcn_wmma_f32_16x16x32_bf16(false, a1[cur].v, false, bb.v, (short)0, acc1, false, false);
      }
      FragU ca0, ca1, bb;
      const uint4* pb = (const uint4*)(wihF + ntg * 256 + lane * 8);
      bb.q[0] = pb[0]; bb.q[1] = pb[1];
      ca0.q[0] = *(const uint4*)&sCtxFrag[0][lane * 8];
      ca0.q[1] = *(const uint4*)&sCtxFrag[0][lane * 8 + 4];
      ca1.q[0] = *(const uint4*)&sCtxFrag[1][lane * 8];
      ca1.q[1] = *(const uint4*)&sCtxFrag[1][lane * 8 + 4];
      acc0 = __builtin_amdgcn_wmma_f32_16x16x32_bf16(false, ca0.v, false, bb.v, (short)0, acc0, false, false);
      acc1 = __builtin_amdgcn_wmma_f32_16x16x32_bf16(false, ca1.v, false, bb.v, (short)0, acc1, false, false);
    }
    __syncthreads();   // all waves done reading U.aFrag before overlaying with gates

    // bias + spill accumulators (C layout: N=lane&15, M=r+8*(lane>=16)) to LDS
    {
      int nCol = ntg * 16 + (lane & 15);
      float bsum = bih[nCol] + bhh[nCol];
      int half = lane >> 4;
      int jloc = (w & 1) * 16 + (lane & 15);
      for (int r = 0; r < 8; ++r) {
        int m = r + 8 * half;
        U.p3.gates[q][m][jloc]      = acc0[r] + bsum;
        U.p3.gates[q][16 + m][jloc] = acc1[r] + bsum;
      }
    }
    __syncthreads();
    // LSTM cell (c stays resident in LDS)
    for (int i = t; i < L * L; i += 256) {
      int l = i >> 5, j = i & 31;
      float ig = sigmoidf_(U.p3.gates[0][l][j]);
      float fg = sigmoidf_(U.p3.gates[1][l][j]);
      float gg = tanhf(U.p3.gates[2][l][j]);
      float og = sigmoidf_(U.p3.gates[3][l][j]);
      float cn = fg * sC[l][j] + ig * gg;
      float hn = og * tanhf(cn);
      sC[l][j] = cn;
      U.p3.hbuf[l][j] = hn;
    }
    __syncthreads();
    // output partial (deterministic; reduced later) + publish h for next step
    const int parn = (s + 1) & 1;
    unsigned short* hPn = hPlain + parn * (L * H);
    unsigned*       hFn = hFrag + parn * (2 * KT * 256);
    if (t < L) {
      float acc2 = 0.f;
      for (int j = 0; j < L; ++j) acc2 += U.p3.hbuf[t][j] * w2[g * L + j];
      partial[(b * L + t) * G + g] = acc2;
    }
    for (int i = t; i < L * L; i += 256) {
      int l = i >> 5, j = i & 31;
      hPn[l * H + g * L + j] = f2bf(U.p3.hbuf[l][j]);
    }
    for (int u = t; u < 2 * 256; u += 256) {   // directly in A-fragment order (our kt == g)
      int mt = u >> 8, r = u & 255;
      int ln = r >> 3, v = r & 7;
      int m = 16 * mt + (ln & 15);
      int k0 = fragK0(v, ln >> 4);
      hFn[(mt * KT + g) * 256 + r] =
          (unsigned)f2bf(U.p3.hbuf[m][k0]) | ((unsigned)f2bf(U.p3.hbuf[m][k0 + 1]) << 16);
    }
    // device-wide step barrier (16 resident WGs; release/acquire through L2)
    __threadfence();
    __syncthreads();
    if (t == 0) {
      atomicAdd(ctr, 1u);
      unsigned target = (unsigned)(G * (s + 1));
      while (__hip_atomic_load(ctr, __ATOMIC_ACQUIRE, __HIP_MEMORY_SCOPE_AGENT) < target)
        __builtin_amdgcn_s_sleep(2);
    }
    __syncthreads();
  }
}

// ---------------- output projection reduce (deterministic) ----------------
__global__ void k_reduce(const float* __restrict__ partial, const float* __restrict__ mask,
                         const float* __restrict__ c2b, float* __restrict__ out) {
  int i = blockIdx.x * blockDim.x + threadIdx.x;
  if (i >= BATCH * L) return;
  float s = c2b[0];
  for (int g = 0; g < G; ++g) s += partial[(size_t)i * G + g];
  out[i] = s * mask[i];
}

extern "C" void kernel_launch(void* const* d_in, const int* in_sizes, int n_in,
                              void* d_out, int out_size, void* d_ws, size_t ws_size,
                              hipStream_t stream) {
  const float* input = (const float*)d_in[0];
  const float* mask  = (const float*)d_in[1];
  const float* biasM = (const float*)d_in[2];
  const float* c1w   = (const float*)d_in[3];
  const float* c1b   = (const float*)d_in[4];
  const float* cw    = (const float*)d_in[5];
  const float* cb    = (const float*)d_in[6];
  const float* fc1w  = (const float*)d_in[7];
  const float* fc1b  = (const float*)d_in[8];
  const float* Wih   = (const float*)d_in[9];
  const float* Whh   = (const float*)d_in[10];
  const float* bih   = (const float*)d_in[11];
  const float* bhh   = (const float*)d_in[12];
  const float* c2w   = (const float*)d_in[13];
  const float* c2b   = (const float*)d_in[14];
  float* out = (float*)d_out;

  char* ws = (char*)d_ws;
  size_t o = 0;
  auto alloc = [&](size_t bytes) { char* p = ws + o; o = (o + bytes + 255) & ~(size_t)255; return p; };
  float*          xF32    = (float*)alloc((size_t)BATCH * L * F * 4);
  float*          uA      = (float*)alloc((size_t)BATCH * F * 4);
  unsigned*       whhF    = (unsigned*)alloc((size_t)KT * NT_ALL * 256 * 4);
  unsigned*       wihF    = (unsigned*)alloc((size_t)NT_ALL * 256 * 4);
  unsigned*       hFrag   = (unsigned*)alloc((size_t)2 * 2 * KT * 256 * 4);
  unsigned short* hPlain  = (unsigned short*)alloc((size_t)2 * L * H * 2);
  float*          partial = (float*)alloc((size_t)BATCH * L * G * 4);
  unsigned*       ctr     = (unsigned*)alloc(256);

  k_pre<<<BATCH, 256, 0, stream>>>(input, mask, c1w, c1b, cw, cb, xF32, uA);
  k_pack<<<KT * NT_ALL + NT_ALL, 256, 0, stream>>>(Whh, Wih, whhF, wihF);
  k_init<<<64, 256, 0, stream>>>(hFrag, (unsigned*)hPlain, ctr);
  k_scan<<<G, 256, 0, stream>>>(uA, xF32, biasM, fc1w, fc1b, whhF, wihF,
                                bih, bhh, c2w, hFrag, hPlain, partial, ctr);
  k_reduce<<<(BATCH * L + 255) / 256, 256, 0, stream>>>(partial, mask, c2b, out);
}
